// MultiHeadAttention_10900626998084
// MI455X (gfx1250) — matrix operand order repaired and checked by reading, not compiled, not run on hardware
//
#include <hip/hip_runtime.h>
#include <hip/hip_bf16.h>

typedef __bf16 bf16_t;
typedef __attribute__((ext_vector_type(16))) __bf16 bf16x16;
typedef __attribute__((ext_vector_type(8)))  float  floatx8;

static constexpr int Bn = 4, Sn = 2048, HIDn = 768, Hn = 12, Dn = 64;
static constexpr float SCALE = 0.125f; // 1/sqrt(64)

__device__ __forceinline__ int lane_id() { return threadIdx.x & 31; }
__device__ __forceinline__ int wave_id() { return threadIdx.x >> 5; }

__device__ __forceinline__ floatx8 wmma_bf16(bf16x16 a, bf16x16 b, floatx8 c) {
  return __builtin_amdgcn_wmma_f32_16x16x32_bf16(false, a, false, b, (short)0, c,
                                                 false, false);
}

// A fragment (16x32, bf16) built from fp32 row-major source.
// element i -> k = k0 + 16*(i/8) + 8*hi + (i&7)
__device__ __forceinline__ bf16x16 load_a_f32(const float* __restrict__ A, int lda,
                                              int row, int k0, int lane) {
  const int hi = (lane >> 4) & 1;
  const float* p = A + (size_t)row * lda + k0 + hi * 8;
  bf16x16 a;
#pragma unroll
  for (int i = 0; i < 8; ++i) { a[i] = (bf16_t)p[i]; a[8 + i] = (bf16_t)p[16 + i]; }
  return a;
}

// ---------------------------------------------------------------------------
// Generic WMMA GEMM: C[M,N] = A[M,K] @ W[K,N] + bias.
// 8 waves/block share nTile -> W tile staged once in LDS per K-step.
// OUT_MODE 0: write bf16 into head-major [B,H,S,D] workspace (for QKV proj)
// OUT_MODE 1: write fp32 row-major [M,N] (final output projection)
// ---------------------------------------------------------------------------
template <bool A_BF16, int OUT_MODE>
__global__ void __launch_bounds__(256)
gemm_wmma_kernel(const void* __restrict__ Ap,
                 const float* __restrict__ W,
                 const float* __restrict__ bias,
                 void* __restrict__ outp,
                 int M, int N, int K) {
  __shared__ bf16_t wtile[32 * 16]; // 1 KB: 32(k) x 16(n) bf16
  const int lane = lane_id();
  const int w = wave_id();
  const int hi = (lane >> 4) & 1;
  const int ln = lane & 15;
  const int nTile = blockIdx.x;
  const int mTile = blockIdx.y * 8 + w;
  const int mBase = mTile * 16;
  const int nBase = nTile * 16;
  const int rowA = mBase + ln;
  const int col = nBase + ln;

  // cooperative W-tile load indices: 512 elems / 256 threads = 2 each
  const int e = threadIdx.x * 2;
  const int kr = e >> 4, cc = e & 15;

  floatx8 c = {};
  for (int k0 = 0; k0 < K; k0 += 32) {
    const float* wp = W + (size_t)(k0 + kr) * N + nBase + cc;
    wtile[e] = (bf16_t)wp[0];
    wtile[e + 1] = (bf16_t)wp[1];
    __syncthreads();

    bf16x16 a;
    if (A_BF16) {
      const bf16_t* A = (const bf16_t*)Ap;
      const bf16_t* p = A + (size_t)rowA * K + k0 + hi * 8;
#pragma unroll
      for (int i = 0; i < 8; ++i) { a[i] = p[i]; a[8 + i] = p[16 + i]; }
    } else {
      a = load_a_f32((const float*)Ap, K, rowA, k0, lane);
    }
    // B fragment from LDS: element i -> k = 16*(i/8) + 8*hi + (i&7), n = ln
    bf16x16 b;
    const bf16_t* bp = wtile + (hi * 8) * 16 + ln;
#pragma unroll
    for (int i = 0; i < 8; ++i) { b[i] = bp[i * 16]; b[8 + i] = bp[(16 + i) * 16]; }
    c = wmma_bf16(a, b, c);
    __syncthreads();
  }

  const float bn = bias[col];
  if (OUT_MODE == 0) {
    bf16_t* dst = (bf16_t*)outp;
    const int h = col >> 6, d = col & 63;
#pragma unroll
    for (int j = 0; j < 8; ++j) {
      const int m = mBase + j + 8 * hi;
      const int bb = m >> 11, s = m & (Sn - 1); // S == 2048
      dst[(((size_t)bb * Hn + h) * Sn + s) * Dn + d] = (bf16_t)(c[j] + bn);
    }
  } else {
    float* dst = (float*)outp;
#pragma unroll
    for (int j = 0; j < 8; ++j) {
      const int m = mBase + j + 8 * hi;
      dst[(size_t)m * N + col] = c[j] + bn;
    }
  }
}

// ---------------------------------------------------------------------------
// Attention: scores = (q @ k^T)*SCALE, mask, softmax -> attn_prob (fp32, once).
// grid (S/16, B*H), 512 threads (16 waves). Wave w owns key tiles [w*8, w*8+8).
// All 2048 scores per query row live in registers (8 x v8f per wave) — no spill.
// ---------------------------------------------------------------------------
__global__ void __launch_bounds__(512)
attn_softmax_kernel(const bf16_t* __restrict__ qb,
                    const bf16_t* __restrict__ kb,
                    const unsigned char* __restrict__ mask,
                    float* __restrict__ attn) {
  __shared__ float red[16][16];
  const int lane = lane_id();
  const int w = wave_id();
  const int hi = (lane >> 4) & 1;
  const int ln = lane & 15;
  const int bh = blockIdx.y;
  const int bidx = bh / Hn;
  const int qTile = blockIdx.x;

  const bf16_t* qh = qb + (size_t)bh * Sn * Dn;
  const bf16_t* kh = kb + (size_t)bh * Sn * Dn;
  float* out = attn + (size_t)bh * Sn * Sn + (size_t)qTile * 16 * Sn;
  const unsigned char* mk = mask + ((size_t)bidx * Sn + qTile * 16) * Sn;

  // Q fragments for k-dim 0..31 and 32..63, reused across all key tiles
  bf16x16 a0, a1;
  {
    const bf16_t* p = qh + (size_t)(qTile * 16 + ln) * Dn + hi * 8;
#pragma unroll
    for (int i = 0; i < 8; ++i) {
      a0[i] = p[i];       a0[8 + i] = p[16 + i];
      a1[i] = p[32 + i];  a1[8 + i] = p[48 + i];
    }
  }

  floatx8 acc[8];
#pragma unroll
  for (int t = 0; t < 8; ++t) {
    const int kTile = w * 8 + t;
    // B operand: B(d, n) = k[n, d]; per-lane table order in d (same as A)
    const bf16_t* pb = kh + (size_t)(kTile * 16 + ln) * Dn + hi * 8;
    bf16x16 b0, b1;
#pragma unroll
    for (int i = 0; i < 8; ++i) { b0[i] = pb[i]; b0[8 + i] = pb[16 + i]; b1[i] = pb[32 + i]; b1[8 + i] = pb[48 + i]; }
    floatx8 c = {};
    c = wmma_bf16(a0, b0, c);
    c = wmma_bf16(a1, b1, c);
    const int col = kTile * 16 + ln;
#pragma unroll
    for (int j = 0; j < 8; ++j) {
      const int row = j + 8 * hi;
      float v = c[j] * SCALE;
      if (mk[(size_t)row * Sn + col]) v = -1e9f;
      acc[t][j] = v;
    }
  }

  // row max: registers -> 16-lane shfl reduce -> cross-wave LDS reduce
  float rowmax[8];
#pragma unroll
  for (int j = 0; j < 8; ++j) {
    float m = -3.0e38f;
#pragma unroll
    for (int t = 0; t < 8; ++t) m = fmaxf(m, acc[t][j]);
#pragma unroll
    for (int off = 1; off < 16; off <<= 1) m = fmaxf(m, __shfl_xor(m, off, 32));
    if (ln == 0) red[w][hi * 8 + j] = m;
  }
  __syncthreads();
#pragma unroll
  for (int j = 0; j < 8; ++j) {
    float m = -3.0e38f;
#pragma unroll
    for (int wv = 0; wv < 16; ++wv) m = fmaxf(m, red[wv][hi * 8 + j]);
    rowmax[j] = m;
  }
  __syncthreads();

  // exp + row sum
  float rinv[8];
#pragma unroll
  for (int j = 0; j < 8; ++j) {
    float s = 0.f;
#pragma unroll
    for (int t = 0; t < 8; ++t) {
      const float e = __expf(acc[t][j] - rowmax[j]);
      acc[t][j] = e;
      s += e;
    }
#pragma unroll
    for (int off = 1; off < 16; off <<= 1) s += __shfl_xor(s, off, 32);
    if (ln == 0) red[w][hi * 8 + j] = s;
  }
  __syncthreads();
#pragma unroll
  for (int j = 0; j < 8; ++j) {
    float s = 0.f;
#pragma unroll
    for (int wv = 0; wv < 16; ++wv) s += red[wv][hi * 8 + j];
    rinv[j] = 1.0f / s;
  }

  // single write of attn_prob
#pragma unroll
  for (int t = 0; t < 8; ++t) {
    const int col = (w * 8 + t) * 16 + ln;
#pragma unroll
    for (int j = 0; j < 8; ++j)
      out[(size_t)(j + 8 * hi) * Sn + col] = acc[t][j] * rinv[j];
  }
}

// ---------------------------------------------------------------------------
// Context: ctx[s,:] = attn_prob[s,:] @ V  (per b,h). V tiles staged in LDS.
// grid (S/128, B*H), 8 waves/block share each 32x64 V tile.
// ---------------------------------------------------------------------------
__global__ void __launch_bounds__(256)
context_kernel(const float* __restrict__ attn,
               const bf16_t* __restrict__ vb,
               bf16_t* __restrict__ ctx) {
  __shared__ __align__(16) bf16_t vtile[32 * 64];
  const int lane = lane_id();
  const int w = wave_id();
  const int hi = (lane >> 4) & 1;
  const int ln = lane & 15;
  const int bh = blockIdx.y;
  const int bidx = bh / Hn, h = bh % Hn;
  const int mTile = blockIdx.x * 8 + w;

  const float*  ah = attn + (size_t)bh * Sn * Sn;
  const bf16_t* vh = vb + (size_t)bh * Sn * Dn;
  const int rowA = mTile * 16 + ln;
  const int e0 = threadIdx.x * 8; // 8 bf16 = 16B per thread per stage

  floatx8 acc[4] = {};
  for (int k0 = 0; k0 < Sn; k0 += 32) {
    *(uint4*)(vtile + e0) = *(const uint4*)(vh + (size_t)k0 * Dn + e0);
    __syncthreads();
    bf16x16 a = load_a_f32(ah, Sn, rowA, k0, lane); // attn fp32 -> bf16
#pragma unroll
    for (int n = 0; n < 4; ++n) {
      const bf16_t* p = vtile + (hi * 8) * 64 + n * 16 + ln;
      bf16x16 b;
#pragma unroll
      for (int i = 0; i < 8; ++i) { b[i] = p[i * 64]; b[8 + i] = p[(16 + i) * 64]; }
      acc[n] = wmma_bf16(a, b, acc[n]);
    }
    __syncthreads();
  }

  // store ctx as [B,S,HID] bf16, col = h*64 + n*16 + ln
#pragma unroll
  for (int n = 0; n < 4; ++n) {
#pragma unroll
    for (int j = 0; j < 8; ++j) {
      const int s = mTile * 16 + j + 8 * hi;
      ctx[((size_t)bidx * Sn + s) * HIDn + h * 64 + n * 16 + ln] = (bf16_t)acc[n][j];
    }
  }
}

extern "C" void kernel_launch(void* const* d_in, const int* in_sizes, int n_in,
                              void* d_out, int out_size, void* d_ws, size_t ws_size,
                              hipStream_t stream) {
  (void)in_sizes; (void)n_in; (void)out_size; (void)ws_size;
  const float* Q  = (const float*)d_in[0];
  const float* K_ = (const float*)d_in[1];
  const float* V  = (const float*)d_in[2];
  const unsigned char* mask = (const unsigned char*)d_in[3];
  const float* Wq = (const float*)d_in[4];
  const float* bq = (const float*)d_in[5];
  const float* Wk = (const float*)d_in[6];
  const float* bk = (const float*)d_in[7];
  const float* Wv = (const float*)d_in[8];
  const float* bv = (const float*)d_in[9];
  const float* Wo = (const float*)d_in[10];
  const float* bo = (const float*)d_in[11];

  float* out  = (float*)d_out;
  float* attn = out + (size_t)Bn * Sn * HIDn; // outputs concatenated: output, attn_prob

  const size_t NE = (size_t)Bn * Sn * HIDn; // 6,291,456 elements
  bf16_t* qb  = (bf16_t*)d_ws;
  bf16_t* kb  = qb + NE;
  bf16_t* vbp = kb + NE;
  bf16_t* ctx = vbp + NE;

  dim3 gg(HIDn / 16, (Bn * Sn) / 128);
  gemm_wmma_kernel<false, 0><<<gg, 256, 0, stream>>>(Q,  Wq, bq, qb,  Bn * Sn, HIDn, HIDn);
  gemm_wmma_kernel<false, 0><<<gg, 256, 0, stream>>>(K_, Wk, bk, kb,  Bn * Sn, HIDn, HIDn);
  gemm_wmma_kernel<false, 0><<<gg, 256, 0, stream>>>(V,  Wv, bv, vbp, Bn * Sn, HIDn, HIDn);

  attn_softmax_kernel<<<dim3(Sn / 16, Bn * Hn), 512, 0, stream>>>(qb, kb, mask, attn);
  context_kernel<<<dim3(Sn / 128, Bn * Hn), 256, 0, stream>>>(attn, vbp, ctx);

  gemm_wmma_kernel<true, 1><<<gg, 256, 0, stream>>>(ctx, Wo, bo, out, Bn * Sn, HIDn, HIDn);
}
